// Cait_78984448573778
// MI455X (gfx1250) — compile-verified, matrix operand order checked by reading, not run on hardware
//
#include <hip/hip_runtime.h>
#include <cmath>

// ---------------- model constants ----------------
#define BATCH    16
#define CDIM     768
#define NHEADS   12
#define NPATCH   196
#define SEQ2     197          // 1 + NPATCH
#define SPITCH   224          // padded attention K (7 * 32)
#define NDEPTH   12
#define NDEPTH2  2
#define NCLS     1000
#define MROWS    (BATCH * NPATCH)      // 3136
#define ATTN_SCALE 0.125f

// ---------------- WMMA types ----------------
typedef __bf16 bf16_t;
typedef bf16_t bf16x8  __attribute__((ext_vector_type(8)));
typedef bf16_t bf16x16 __attribute__((ext_vector_type(16)));
typedef float  f32x8   __attribute__((ext_vector_type(8)));

#define WMMA_BF16(a, b, c) \
  __builtin_amdgcn_wmma_f32_16x16x32_bf16(false, (a), false, (b), (short)0, (c), false, false)

// ---------------- generic WMMA GEMM: C = epi(A[M,K] * W[N,K]^T) ----------------
// A, W fp32 in memory; converted to bf16 while staging into LDS.
#define GBM 128
#define GBN 128
#define LDP 40   // LDS row pitch in bf16 elems (32 data + 8 pad -> conflict-free 16B reads)

#define EPI_BIAS  1
#define EPI_GELU  2
#define EPI_RESID 4

__device__ __forceinline__ float4 g_ld4(const float* __restrict__ base, int row, int nrows,
                                        int ld, int k) {
  float4 z; z.x = z.y = z.z = z.w = 0.f;
  if (row < nrows) z = *(const float4*)(base + (size_t)row * ld + k);
  return z;
}

__device__ __forceinline__ void st_lds(bf16_t* t, int slot, float4 v) {
  int r = slot >> 3, k = (slot & 7) << 2;
  bf16_t* p = t + r * LDP + k;
  p[0] = (bf16_t)v.x; p[1] = (bf16_t)v.y; p[2] = (bf16_t)v.z; p[3] = (bf16_t)v.w;
}

// 16-bit A/B fragment per ISA layout: lane = row, half = lane>>4 selects K chunks
// [8*half, 8*half+8) and [16+8*half, 16+8*half+8)  -> two 16-byte LDS loads.
__device__ __forceinline__ bf16x16 lds_frag(const bf16_t* t, int row0, int lane) {
  int r = lane & 15, h = lane >> 4;
  const bf16_t* p = t + (row0 + r) * LDP + 8 * h;
  bf16x8 lo = *(const bf16x8*)p;
  bf16x8 hi = *(const bf16x8*)(p + 16);
  return __builtin_shufflevector(lo, hi, 0,1,2,3,4,5,6,7,8,9,10,11,12,13,14,15);
}

__global__ __launch_bounds__(256)
void wmma_gemm_k(const float* __restrict__ A, const float* __restrict__ W,
                 const float* __restrict__ bias, float* __restrict__ Co,
                 const float* __restrict__ resid, const float* __restrict__ lscale,
                 int M, int N, int K, int flags) {
  __shared__ __align__(16) bf16_t As[2][GBM * LDP];
  __shared__ __align__(16) bf16_t Bs[2][GBN * LDP];

  const int tid = threadIdx.x, lane = tid & 31, wid = tid >> 5;
  const int wm = wid & 3, wn = wid >> 2;        // 4x2 wave grid: 32 rows x 64 cols each
  const int mb = blockIdx.y * GBM, nb = blockIdx.x * GBN;
  const int steps = K >> 5;

  f32x8 acc[2][4];
#pragma unroll
  for (int i = 0; i < 2; i++)
#pragma unroll
    for (int j = 0; j < 4; j++)
#pragma unroll
      for (int e = 0; e < 8; e++) acc[i][j][e] = 0.f;

  float4 ra[4], rb[4];
#pragma unroll
  for (int i = 0; i < 4; i++) {
    int s = tid + (i << 8);
    ra[i] = g_ld4(A, mb + (s >> 3), M, K, ((s & 7) << 2));
    rb[i] = g_ld4(W, nb + (s >> 3), N, K, ((s & 7) << 2));
  }
#pragma unroll
  for (int i = 0; i < 4; i++) { int s = tid + (i << 8); st_lds(As[0], s, ra[i]); st_lds(Bs[0], s, rb[i]); }
  __syncthreads();

  int buf = 0;
  for (int kt = 0; kt < steps; kt++) {
    if (kt + 1 < steps) {
      int k0 = (kt + 1) << 5;
#pragma unroll
      for (int i = 0; i < 4; i++) {
        int s = tid + (i << 8);
        ra[i] = g_ld4(A, mb + (s >> 3), M, K, k0 + ((s & 7) << 2));
        rb[i] = g_ld4(W, nb + (s >> 3), N, K, k0 + ((s & 7) << 2));
      }
    }
    bf16x16 af[2], bfv[4];
#pragma unroll
    for (int im = 0; im < 2; im++) af[im] = lds_frag(As[buf], wm * 32 + im * 16, lane);
#pragma unroll
    for (int in = 0; in < 4; in++) bfv[in] = lds_frag(Bs[buf], wn * 64 + in * 16, lane);
#pragma unroll
    for (int im = 0; im < 2; im++)
#pragma unroll
      for (int in = 0; in < 4; in++)
        acc[im][in] = WMMA_BF16(af[im], bfv[in], acc[im][in]);

    if (kt + 1 < steps) {
      int nbuf = buf ^ 1;
#pragma unroll
      for (int i = 0; i < 4; i++) { int s = tid + (i << 8); st_lds(As[nbuf], s, ra[i]); st_lds(Bs[nbuf], s, rb[i]); }
    }
    __syncthreads();
    buf ^= 1;
  }

  // epilogue: C layout lane 0-15 -> M=r, lane 16-31 -> M=8+r; N = lane&15
  const int half = lane >> 4, ln = lane & 15;
#pragma unroll
  for (int im = 0; im < 2; im++)
#pragma unroll
    for (int in = 0; in < 4; in++) {
      int n = nb + wn * 64 + in * 16 + ln;
      if (n >= N) continue;
      float bv = (flags & EPI_BIAS) ? bias[n] : 0.f;
      float sc = (flags & EPI_RESID) ? lscale[n] : 0.f;
#pragma unroll
      for (int r = 0; r < 8; r++) {
        int m = mb + wm * 32 + im * 16 + half * 8 + r;
        if (m >= M) continue;
        float v = acc[im][in][r] + bv;
        if (flags & EPI_GELU) v = 0.5f * v * (1.f + erff(v * 0.70710678118f));
        if (flags & EPI_RESID) v = resid[(size_t)m * N + n] + sc * v;
        Co[(size_t)m * N + n] = v;
      }
    }
}

// ---------------- attention: build a bf16 fragment straight from a global row ----------------
__device__ __forceinline__ bf16x16 frag_g(const float* __restrict__ p, int h, float scale) {
  const float* a = p + 8 * h;
  float4 x0 = *(const float4*)a;
  float4 x1 = *(const float4*)(a + 4);
  float4 x2 = *(const float4*)(a + 16);
  float4 x3 = *(const float4*)(a + 20);
  bf16x16 r;
  r[0]  = (bf16_t)(x0.x * scale); r[1]  = (bf16_t)(x0.y * scale);
  r[2]  = (bf16_t)(x0.z * scale); r[3]  = (bf16_t)(x0.w * scale);
  r[4]  = (bf16_t)(x1.x * scale); r[5]  = (bf16_t)(x1.y * scale);
  r[6]  = (bf16_t)(x1.z * scale); r[7]  = (bf16_t)(x1.w * scale);
  r[8]  = (bf16_t)(x2.x * scale); r[9]  = (bf16_t)(x2.y * scale);
  r[10] = (bf16_t)(x2.z * scale); r[11] = (bf16_t)(x2.w * scale);
  r[12] = (bf16_t)(x3.x * scale); r[13] = (bf16_t)(x3.y * scale);
  r[14] = (bf16_t)(x3.z * scale); r[15] = (bf16_t)(x3.w * scale);
  return r;
}

// S[b,h,196,224] = scale * q @ k^T  (wave per 16x16 tile; pad cols zeroed)
__global__ __launch_bounds__(256)
void attn_score_k(const float* __restrict__ qkv, float* __restrict__ S) {
  int lane = threadIdx.x & 31, wid = threadIdx.x >> 5;
  int t = blockIdx.x * 8 + wid;
  if (t >= 13 * 14) return;
  int mt = t / 14, nt = t % 14;
  int bh = blockIdx.y, b = bh / NHEADS, h = bh % NHEADS;
  const float* qb = qkv + (size_t)b * NPATCH * (3 * CDIM) + h * 64;
  const float* kb = qb + CDIM;
  int r = lane & 15, hf = lane >> 4;
  int mrow = mt * 16 + r; if (mrow > NPATCH - 1) mrow = NPATCH - 1;
  int nrow = nt * 16 + r; if (nrow > NPATCH - 1) nrow = NPATCH - 1;

  f32x8 acc;
#pragma unroll
  for (int e = 0; e < 8; e++) acc[e] = 0.f;
#pragma unroll
  for (int ks = 0; ks < 64; ks += 32) {
    bf16x16 aq = frag_g(qb + (size_t)mrow * (3 * CDIM) + ks, hf, ATTN_SCALE);
    bf16x16 bk = frag_g(kb + (size_t)nrow * (3 * CDIM) + ks, hf, 1.f);
    acc = WMMA_BF16(aq, bk, acc);
  }
  float* Sb = S + (size_t)bh * NPATCH * SPITCH;
  int nn = nt * 16 + (lane & 15);
#pragma unroll
  for (int rr = 0; rr < 8; rr++) {
    int mm = mt * 16 + hf * 8 + rr;
    if (mm < NPATCH && nn < SPITCH)
      Sb[(size_t)mm * SPITCH + nn] = (nn < NPATCH) ? acc[rr] : 0.f;
  }
}

// O[b,n, h*64+d] = attn[b,h,n,:] @ v   (K = 224 padded, v guarded to 0)
__global__ __launch_bounds__(256)
void attn_av_k(const float* __restrict__ Sp, const float* __restrict__ qkv,
               float* __restrict__ O) {
  int lane = threadIdx.x & 31, wid = threadIdx.x >> 5;
  int t = blockIdx.x * 8 + wid;
  if (t >= 13 * 4) return;
  int mt = t >> 2, dt = t & 3;
  int bh = blockIdx.y, b = bh / NHEADS, h = bh % NHEADS;
  const float* Ab = Sp + (size_t)bh * NPATCH * SPITCH;
  const float* vb = qkv + (size_t)b * NPATCH * (3 * CDIM) + 2 * CDIM + h * 64;
  int r = lane & 15, hf = lane >> 4;
  int arow = mt * 16 + r; if (arow > NPATCH - 1) arow = NPATCH - 1;
  int dcol = dt * 16 + r;

  f32x8 acc;
#pragma unroll
  for (int e = 0; e < 8; e++) acc[e] = 0.f;
  for (int ks = 0; ks < SPITCH; ks += 32) {
    bf16x16 af = frag_g(Ab + (size_t)arow * SPITCH + ks, hf, 1.f);
    bf16x16 bv;
#pragma unroll
    for (int j = 0; j < 8; j++) {
      int k1 = ks + 8 * hf + j;
      int k2 = ks + 16 + 8 * hf + j;
      float v1 = (k1 < NPATCH) ? vb[(size_t)k1 * (3 * CDIM) + dcol] : 0.f;
      float v2 = (k2 < NPATCH) ? vb[(size_t)k2 * (3 * CDIM) + dcol] : 0.f;
      bv[j] = (bf16_t)v1; bv[8 + j] = (bf16_t)v2;
    }
    acc = WMMA_BF16(af, bv, acc);
  }
  int dd = dt * 16 + (lane & 15);
#pragma unroll
  for (int rr = 0; rr < 8; rr++) {
    int mm = mt * 16 + hf * 8 + rr;
    if (mm < NPATCH)
      O[((size_t)b * NPATCH + mm) * CDIM + h * 64 + dd] = acc[rr];
  }
}

// ---------------- talking-heads mix: out[b,g,n,m] = sum_h in[b,h,n,m]*w[g,h] + bias[g]
__global__ void headmix_k(const float* __restrict__ in, const float* __restrict__ w,
                          const float* __restrict__ bias, float* __restrict__ out) {
  int idx = blockIdx.x * 256 + threadIdx.x;
  if (idx >= BATCH * NPATCH * NPATCH) return;
  int m = idx % NPATCH; int rem = idx / NPATCH;
  int n = rem % NPATCH; int b = rem / NPATCH;
  size_t base = (((size_t)b * NHEADS) * NPATCH + n) * SPITCH + m;
  size_t hstr = (size_t)NPATCH * SPITCH;
  float v[NHEADS];
#pragma unroll
  for (int hh = 0; hh < NHEADS; hh++) v[hh] = in[base + hh * hstr];
#pragma unroll
  for (int g = 0; g < NHEADS; g++) {
    float a = bias[g];
#pragma unroll
    for (int hh = 0; hh < NHEADS; hh++) a += w[g * NHEADS + hh] * v[hh];
    out[base + g * hstr] = a;
  }
}

// ---------------- softmax over rows of the pitched attention buffer ----------------
__global__ __launch_bounds__(256)
void softmax_rows_k(float* __restrict__ S, int nrows) {
  int wid = threadIdx.x >> 5, lane = threadIdx.x & 31;
  int row = blockIdx.x * 8 + wid;
  if (row >= nrows) return;
  float* p = S + (size_t)row * SPITCH;
  float v[7]; float mx = -3.4e38f;
#pragma unroll
  for (int i = 0; i < 7; i++) {
    int c = lane + i * 32;
    v[i] = (c < NPATCH) ? p[c] : -3.4e38f;
    mx = fmaxf(mx, v[i]);
  }
#pragma unroll
  for (int m = 16; m; m >>= 1) mx = fmaxf(mx, __shfl_xor(mx, m, 32));
  float s = 0.f;
#pragma unroll
  for (int i = 0; i < 7; i++) {
    int c = lane + i * 32;
    if (c < NPATCH) { v[i] = __expf(v[i] - mx); s += v[i]; }
  }
#pragma unroll
  for (int m = 16; m; m >>= 1) s += __shfl_xor(s, m, 32);
  float inv = 1.f / s;
#pragma unroll
  for (int i = 0; i < 7; i++) {
    int c = lane + i * 32;
    if (c < NPATCH) p[c] = v[i] * inv;
  }
  for (int c = NPATCH + lane; c < SPITCH; c += 32) p[c] = 0.f;  // zero K-pad
}

// ---------------- block reductions ----------------
__device__ __forceinline__ float blk_sum(float v, float* red) {
#pragma unroll
  for (int m = 16; m; m >>= 1) v += __shfl_xor(v, m, 32);
  int tid = threadIdx.x;
  if ((tid & 31) == 0) red[tid >> 5] = v;
  __syncthreads();
  if (tid == 0) { float r = 0.f; for (int i = 0; i < 8; i++) r += red[i]; red[0] = r; }
  __syncthreads();
  float r = red[0];
  __syncthreads();
  return r;
}
__device__ __forceinline__ float blk_max(float v, float* red) {
#pragma unroll
  for (int m = 16; m; m >>= 1) v = fmaxf(v, __shfl_xor(v, m, 32));
  int tid = threadIdx.x;
  if ((tid & 31) == 0) red[tid >> 5] = v;
  __syncthreads();
  if (tid == 0) { float r = red[0]; for (int i = 1; i < 8; i++) r = fmaxf(r, red[i]); red[0] = r; }
  __syncthreads();
  float r = red[0];
  __syncthreads();
  return r;
}

// ---------------- layernorm over C=768, one block per row ----------------
__global__ __launch_bounds__(256)
void ln_k(const float* __restrict__ x, const float* __restrict__ w,
          const float* __restrict__ bb, float* __restrict__ y) {
  __shared__ float red[8];
  int row = blockIdx.x, tid = threadIdx.x;
  const float* xr = x + (size_t)row * CDIM;
  float v0 = xr[tid], v1 = xr[tid + 256], v2 = xr[tid + 512];
  float mean = blk_sum(v0 + v1 + v2, red) * (1.f / CDIM);
  float d0 = v0 - mean, d1 = v1 - mean, d2 = v2 - mean;
  float var = blk_sum(d0 * d0 + d1 * d1 + d2 * d2, red) * (1.f / CDIM);
  float inv = rsqrtf(var + 1e-6f);
  float* yr = y + (size_t)row * CDIM;
  yr[tid]       = d0 * inv * w[tid]       + bb[tid];
  yr[tid + 256] = d1 * inv * w[tid + 256] + bb[tid + 256];
  yr[tid + 512] = d2 * inv * w[tid + 512] + bb[tid + 512];
}

// ---------------- small data-movement kernels ----------------
__global__ void im2row_k(const float* __restrict__ x, float* __restrict__ out) {
  int idx = blockIdx.x * 256 + threadIdx.x;
  if (idx >= MROWS * CDIM) return;
  int col = idx % CDIM, row = idx / CDIM;
  int b = row / NPATCH, pp = row % NPATCH;
  int gh = pp / 14, gw = pp % 14;
  int c3 = col >> 8, rem = col & 255;
  int py = rem >> 4, px = rem & 15;
  out[idx] = x[(((size_t)b * 3 + c3) * 224 + gh * 16 + py) * 224 + gw * 16 + px];
}
__global__ void posadd_k(float* __restrict__ h, const float* __restrict__ pos) {
  int idx = blockIdx.x * 256 + threadIdx.x;
  if (idx >= MROWS * CDIM) return;
  int c = idx % CDIM, row = idx / CDIM, np = row % NPATCH;
  h[idx] += pos[np * CDIM + c];
}
__global__ void initcls_k(const float* __restrict__ ct, float* __restrict__ cls) {
  int i = blockIdx.x * 256 + threadIdx.x;
  if (i < BATCH * CDIM) cls[i] = ct[i % CDIM];
}
__global__ void concat_k(const float* __restrict__ cls, const float* __restrict__ h,
                         float* __restrict__ u) {
  int idx = blockIdx.x * 256 + threadIdx.x;
  if (idx >= BATCH * SEQ2 * CDIM) return;
  int c = idx % CDIM, row = idx / CDIM;
  int b = row / SEQ2, t = row % SEQ2;
  u[idx] = (t == 0) ? cls[b * CDIM + c]
                    : h[((size_t)b * NPATCH + (t - 1)) * CDIM + c];
}
__global__ void gathercls_k(const float* __restrict__ un, float* __restrict__ g) {
  int i = blockIdx.x * 256 + threadIdx.x;
  if (i >= BATCH * CDIM) return;
  int b = i / CDIM, c = i % CDIM;
  g[i] = un[(size_t)b * SEQ2 * CDIM + c];
}

// ---------------- class attention core (tiny: 1 query token) ----------------
__global__ __launch_bounds__(256)
void ca_attn_k(const float* __restrict__ q, const float* __restrict__ kb,
               const float* __restrict__ vb, float* __restrict__ o) {
  __shared__ float sc[SEQ2];
  __shared__ float qs[64];
  __shared__ float red[8];
  int tid = threadIdx.x;
  int b = blockIdx.x / NHEADS, h = blockIdx.x % NHEADS;
  if (tid < 64) qs[tid] = q[b * CDIM + h * 64 + tid] * ATTN_SCALE;
  __syncthreads();
  float val = -3.4e38f;
  if (tid < SEQ2) {
    const float* kr = kb + ((size_t)b * SEQ2 + tid) * CDIM + h * 64;
    float d = 0.f;
#pragma unroll 8
    for (int j = 0; j < 64; j++) d += qs[j] * kr[j];
    sc[tid] = d; val = d;
  }
  float mx = blk_max(val, red);
  float e = 0.f;
  if (tid < SEQ2) { e = __expf(sc[tid] - mx); sc[tid] = e; }
  float s = blk_sum(e, red);
  float inv = 1.f / s;
  if (tid < 64) {
    float a = 0.f;
    for (int m = 0; m < SEQ2; m++)
      a += sc[m] * vb[((size_t)b * SEQ2 + m) * CDIM + h * 64 + tid];
    o[b * CDIM + h * 64 + tid] = a * inv;
  }
}

// ---------------- host orchestration ----------------
static inline dim3 gemm_grid(int M, int N) {
  return dim3((unsigned)((N + GBN - 1) / GBN), (unsigned)((M + GBM - 1) / GBM));
}

extern "C" void kernel_launch(void* const* d_in, const int* in_sizes, int n_in,
                              void* d_out, int out_size, void* d_ws, size_t ws_size,
                              hipStream_t stream) {
  const float* x        = (const float*)d_in[0];
  const float* patch_w  = (const float*)d_in[1];
  const float* patch_b  = (const float*)d_in[2];
  const float* cls_tok  = (const float*)d_in[3];
  const float* pos_emb  = (const float*)d_in[4];
  const float* n1w = (const float*)d_in[5],  *n1b = (const float*)d_in[6];
  const float* qkvw = (const float*)d_in[7], *qkvb = (const float*)d_in[8];
  const float* plw = (const float*)d_in[9],  *plb = (const float*)d_in[10];
  const float* pww = (const float*)d_in[11], *pwb = (const float*)d_in[12];
  const float* projw = (const float*)d_in[13], *projb = (const float*)d_in[14];
  const float* n2w = (const float*)d_in[15], *n2b = (const float*)d_in[16];
  const float* f1w = (const float*)d_in[17], *f1b = (const float*)d_in[18];
  const float* f2w = (const float*)d_in[19], *f2b = (const float*)d_in[20];
  const float* g1 = (const float*)d_in[21],  *g2 = (const float*)d_in[22];
  const float* tn1w = (const float*)d_in[23], *tn1b = (const float*)d_in[24];
  const float* tqw = (const float*)d_in[25], *tqb = (const float*)d_in[26];
  const float* tkw = (const float*)d_in[27], *tkb = (const float*)d_in[28];
  const float* tvw = (const float*)d_in[29], *tvb = (const float*)d_in[30];
  const float* tprojw = (const float*)d_in[31], *tprojb = (const float*)d_in[32];
  const float* tn2w = (const float*)d_in[33], *tn2b = (const float*)d_in[34];
  const float* tf1w = (const float*)d_in[35], *tf1b = (const float*)d_in[36];
  const float* tf2w = (const float*)d_in[37], *tf2b = (const float*)d_in[38];
  const float* tg1 = (const float*)d_in[39], *tg2 = (const float*)d_in[40];
  const float* normw = (const float*)d_in[41], *normb = (const float*)d_in[42];
  const float* headw = (const float*)d_in[43], *headb = (const float*)d_in[44];
  float* out = (float*)d_out;
  (void)in_sizes; (void)n_in; (void)out_size; (void)ws_size;

  // workspace carve (256B aligned)
  char* base = (char*)d_ws;
  size_t off = 0;
  auto carve = [&](size_t bytes) -> float* {
    float* p = (float*)(base + off);
    off = (off + bytes + 255) & ~(size_t)255;
    return p;
  };
  float* h      = carve((size_t)MROWS * CDIM * 4);
  float* lnbuf  = carve((size_t)MROWS * CDIM * 4);
  float* qkv    = carve((size_t)MROWS * 3 * CDIM * 4);
  float* Sa     = carve((size_t)BATCH * NHEADS * NPATCH * SPITCH * 4);
  float* Sb     = carve((size_t)BATCH * NHEADS * NPATCH * SPITCH * 4);
  float* obuf   = carve((size_t)MROWS * CDIM * 4);
  float* hid    = carve((size_t)MROWS * 4 * CDIM * 4);   // MLP hidden; also im2row alias
  float* u      = carve((size_t)BATCH * SEQ2 * CDIM * 4);
  float* un     = carve((size_t)BATCH * SEQ2 * CDIM * 4);
  float* kbuf   = carve((size_t)BATCH * SEQ2 * CDIM * 4);
  float* vbuf   = carve((size_t)BATCH * SEQ2 * CDIM * 4);
  float* clsbuf = carve((size_t)BATCH * CDIM * 4);
  float* clsg   = carve((size_t)BATCH * CDIM * 4);
  float* clsq   = carve((size_t)BATCH * CDIM * 4);
  float* clso   = carve((size_t)BATCH * CDIM * 4);
  float* clsln  = carve((size_t)BATCH * CDIM * 4);
  float* clshid = carve((size_t)BATCH * 4 * CDIM * 4);
  float* im2row = hid;  // alias: disjoint lifetime

  const int ECNT = MROWS * CDIM;
  const dim3 B256(256);

  // ---- patch embed ----
  im2row_k<<<(ECNT + 255) / 256, B256, 0, stream>>>(x, im2row);
  wmma_gemm_k<<<gemm_grid(MROWS, CDIM), B256, 0, stream>>>(
      im2row, patch_w, patch_b, h, nullptr, nullptr, MROWS, CDIM, CDIM, EPI_BIAS);
  posadd_k<<<(ECNT + 255) / 256, B256, 0, stream>>>(h, pos_emb);

  // ---- talking-head blocks ----
  const int mixN = BATCH * NPATCH * NPATCH;
  const int smRows = BATCH * NHEADS * NPATCH;
  for (int l = 0; l < NDEPTH; l++) {
    ln_k<<<MROWS, B256, 0, stream>>>(h, n1w + l * CDIM, n1b + l * CDIM, lnbuf);
    wmma_gemm_k<<<gemm_grid(MROWS, 3 * CDIM), B256, 0, stream>>>(
        lnbuf, qkvw + (size_t)l * 3 * CDIM * CDIM, qkvb + l * 3 * CDIM, qkv,
        nullptr, nullptr, MROWS, 3 * CDIM, CDIM, EPI_BIAS);
    attn_score_k<<<dim3(23, BATCH * NHEADS), B256, 0, stream>>>(qkv, Sa);
    headmix_k<<<(mixN + 255) / 256, B256, 0, stream>>>(
        Sa, plw + l * NHEADS * NHEADS, plb + l * NHEADS, Sb);
    softmax_rows_k<<<(smRows + 7) / 8, B256, 0, stream>>>(Sb, smRows);
    headmix_k<<<(mixN + 255) / 256, B256, 0, stream>>>(
        Sb, pww + l * NHEADS * NHEADS, pwb + l * NHEADS, Sa);
    attn_av_k<<<dim3(7, BATCH * NHEADS), B256, 0, stream>>>(Sa, qkv, obuf);
    wmma_gemm_k<<<gemm_grid(MROWS, CDIM), B256, 0, stream>>>(
        obuf, projw + (size_t)l * CDIM * CDIM, projb + l * CDIM, h,
        h, g1 + l * CDIM, MROWS, CDIM, CDIM, EPI_BIAS | EPI_RESID);
    ln_k<<<MROWS, B256, 0, stream>>>(h, n2w + l * CDIM, n2b + l * CDIM, lnbuf);
    wmma_gemm_k<<<gemm_grid(MROWS, 4 * CDIM), B256, 0, stream>>>(
        lnbuf, f1w + (size_t)l * 4 * CDIM * CDIM, f1b + l * 4 * CDIM, hid,
        nullptr, nullptr, MROWS, 4 * CDIM, CDIM, EPI_BIAS | EPI_GELU);
    wmma_gemm_k<<<gemm_grid(MROWS, CDIM), B256, 0, stream>>>(
        hid, f2w + (size_t)l * CDIM * 4 * CDIM, f2b + l * CDIM, h,
        h, g2 + l * CDIM, MROWS, CDIM, 4 * CDIM, EPI_BIAS | EPI_RESID);
  }

  // ---- class-attention blocks ----
  initcls_k<<<(BATCH * CDIM + 255) / 256, B256, 0, stream>>>(cls_tok, clsbuf);
  const int uN = BATCH * SEQ2 * CDIM;
  for (int l = 0; l < NDEPTH2; l++) {
    concat_k<<<(uN + 255) / 256, B256, 0, stream>>>(clsbuf, h, u);
    ln_k<<<BATCH * SEQ2, B256, 0, stream>>>(u, tn1w + l * CDIM, tn1b + l * CDIM, un);
    gathercls_k<<<(BATCH * CDIM + 255) / 256, B256, 0, stream>>>(un, clsg);
    wmma_gemm_k<<<gemm_grid(BATCH, CDIM), B256, 0, stream>>>(
        clsg, tqw + (size_t)l * CDIM * CDIM, tqb + l * CDIM, clsq,
        nullptr, nullptr, BATCH, CDIM, CDIM, EPI_BIAS);
    wmma_gemm_k<<<gemm_grid(BATCH * SEQ2, CDIM), B256, 0, stream>>>(
        un, tkw + (size_t)l * CDIM * CDIM, tkb + l * CDIM, kbuf,
        nullptr, nullptr, BATCH * SEQ2, CDIM, CDIM, EPI_BIAS);
    wmma_gemm_k<<<gemm_grid(BATCH * SEQ2, CDIM), B256, 0, stream>>>(
        un, tvw + (size_t)l * CDIM * CDIM, tvb + l * CDIM, vbuf,
        nullptr, nullptr, BATCH * SEQ2, CDIM, CDIM, EPI_BIAS);
    ca_attn_k<<<BATCH * NHEADS, B256, 0, stream>>>(clsq, kbuf, vbuf, clso);
    wmma_gemm_k<<<gemm_grid(BATCH, CDIM), B256, 0, stream>>>(
        clso, tprojw + (size_t)l * CDIM * CDIM, tprojb + l * CDIM, clsbuf,
        clsbuf, tg1 + l * CDIM, BATCH, CDIM, CDIM, EPI_BIAS | EPI_RESID);
    ln_k<<<BATCH, B256, 0, stream>>>(clsbuf, tn2w + l * CDIM, tn2b + l * CDIM, clsln);
    wmma_gemm_k<<<gemm_grid(BATCH, 4 * CDIM), B256, 0, stream>>>(
        clsln, tf1w + (size_t)l * 4 * CDIM * CDIM, tf1b + l * 4 * CDIM, clshid,
        nullptr, nullptr, BATCH, 4 * CDIM, CDIM, EPI_BIAS | EPI_GELU);
    wmma_gemm_k<<<gemm_grid(BATCH, CDIM), B256, 0, stream>>>(
        clshid, tf2w + (size_t)l * CDIM * 4 * CDIM, tf2b + l * CDIM, clsbuf,
        clsbuf, tg2 + l * CDIM, BATCH, CDIM, 4 * CDIM, EPI_BIAS | EPI_RESID);
  }

  // ---- final LN (only cls row is used) + classifier head ----
  ln_k<<<BATCH, B256, 0, stream>>>(clsbuf, normw, normb, clsln);
  wmma_gemm_k<<<gemm_grid(BATCH, NCLS), B256, 0, stream>>>(
      clsln, headw, headb, out, nullptr, nullptr, BATCH, NCLS, CDIM, EPI_BIAS);
}
